// GeometricAugmentor_78271484002637
// MI455X (gfx1250) — compile-verified, matrix operand order checked by reading, not run on hardware
//
#include <hip/hip_runtime.h>
#include <hip/hip_bf16.h>
#include <math.h>

typedef _Float16 half_t;
typedef __attribute__((ext_vector_type(16))) _Float16 v16h;
typedef __attribute__((ext_vector_type(8)))  _Float16 v8h;
typedef __attribute__((ext_vector_type(8)))  float    v8f;

#define BATCH   64
#define N_TOK   196
#define DIM_    512
#define PD_     768
#define MLP_    2048
#define NPAD    224              // 196 padded to 7*32 for WMMA K
#define M_ALL   (BATCH * N_TOK)  // 12544, divisible by 64
#define LDS_STRIDE 40            // f16 elems per LDS tile row (80B: 16B-aligned,
                                 // 20-bank shift/row -> spread ds_load_b128)

// ---------------------------------------------------------------------------
// WMMA fragment helpers (f16 16x16x32 lane layout, wave32)
// lane = h2*16 + r : A lane r holds row M=r; K chunks [8*h2 .. +7] and
// [16+8*h2 .. +7].  B fragment mirrors this with lane r = column N=r.
// ---------------------------------------------------------------------------
union V16U { v16h v; v8h h[2]; };

__device__ __forceinline__ v16h load_frag(const half_t* p) {
  V16U u;
  u.h[0] = *(const v8h*)(p);
  u.h[1] = *(const v8h*)(p + 16);
  return u.v;
}

// Stage one (ROWS x 32) f16 tile into LDS with CDNA5 async copies.
// Each lane moves CHUNKS 16-byte chunks; chunk c = (row, 8*(c&3)).
// Rows are clamped so OOB tiles re-read the last valid row (masked at store).
template<int CHUNKS>
__device__ __forceinline__ void async_copy_tile(const half_t* __restrict__ gbase,
                                                int ld, int maxRow, int baseRow,
                                                int k0, half_t* sm, int tid)
{
#pragma unroll
  for (int j = 0; j < CHUNKS; ++j) {
    int c   = tid + 128 * j;
    int row = c >> 2;
    int k8  = (c & 3) * 8;
    int gr  = baseRow + row; if (gr > maxRow) gr = maxRow;
    const half_t* gp = gbase + (size_t)gr * ld + (k0 + k8);
    unsigned loff = (unsigned)(uintptr_t)(sm + row * LDS_STRIDE + k8);
    asm volatile("global_load_async_to_lds_b128 %0, %1, off"
                 :: "v"(loff), "v"(gp) : "memory");
  }
}

enum { EPI_F32 = 0, EPI_F16 = 1, EPI_F16_GELU = 2, EPI_F32_BIAS_RES = 3 };

// D = alpha * (A @ Bt^T) + bias (+resid), A:(M,K) f16 row-major, Bt:(N,K) f16
// Block tile 64x128 (4 waves, 32x64 each), K-step 32, double-buffered LDS
// staging via GLOBAL_LOAD_ASYNC_TO_LDS_B128 / s_wait_asynccnt.
__global__ __launch_bounds__(128)
void gemm_wmma(const half_t* __restrict__ A, long long sA, int lda,
               const half_t* __restrict__ Bt, long long sB, int ldb,
               void* __restrict__ Dv, long long sD, int ldd,
               const float* __restrict__ bias,
               const float* __restrict__ resid,
               int M, int Nn, int K, float alpha, int mode)
{
  __shared__ half_t smA[2][64  * LDS_STRIDE];   //  5 KB per buffer
  __shared__ half_t smB[2][128 * LDS_STRIDE];   // 10 KB per buffer
  const int tid  = threadIdx.x;
  const int lane = tid & 31;
  const int wave = tid >> 5;
  const int r  = lane & 15;
  const int h2 = lane >> 4;
  const int wm = (wave >> 1) * 32;   // wave row offset in block tile
  const int wn = (wave & 1)  * 64;   // wave col offset in block tile
  const int baseM = blockIdx.x * 64;
  const int baseN = blockIdx.y * 128;
  const int z = blockIdx.z;
  A  += (size_t)z * (size_t)sA;
  Bt += (size_t)z * (size_t)sB;

  v8f acc[2][4];
#pragma unroll
  for (int mi = 0; mi < 2; ++mi)
#pragma unroll
    for (int ni = 0; ni < 4; ++ni)
      acc[mi][ni] = (v8f){0.f,0.f,0.f,0.f,0.f,0.f,0.f,0.f};

  // prologue: stage first K tile into buffer 0 (6 async ops per wave)
  async_copy_tile<2>(A,  lda, M  - 1, baseM, 0, &smA[0][0], tid);
  async_copy_tile<4>(Bt, ldb, Nn - 1, baseN, 0, &smB[0][0], tid);

  for (int k0 = 0; k0 < K; k0 += 32) {
    const int cur = (k0 >> 5) & 1;
    if (k0 + 32 < K) {
      // issue next tile into the other buffer, then release the current
      // batch: async loads complete in order, so asynccnt<=6 == batch done.
      async_copy_tile<2>(A,  lda, M  - 1, baseM, k0 + 32, &smA[cur ^ 1][0], tid);
      async_copy_tile<4>(Bt, ldb, Nn - 1, baseN, k0 + 32, &smB[cur ^ 1][0], tid);
      asm volatile("s_wait_asynccnt 0x6" ::: "memory");
    } else {
      asm volatile("s_wait_asynccnt 0x0" ::: "memory");
    }
    __syncthreads();   // all waves' tile data visible in LDS

    v16h a[2], b[4];
#pragma unroll
    for (int mi = 0; mi < 2; ++mi)
      a[mi] = load_frag(&smA[cur][(wm + mi * 16 + r) * LDS_STRIDE + 8 * h2]);
#pragma unroll
    for (int ni = 0; ni < 4; ++ni)
      b[ni] = load_frag(&smB[cur][(wn + ni * 16 + r) * LDS_STRIDE + 8 * h2]);
#pragma unroll
    for (int mi = 0; mi < 2; ++mi)
#pragma unroll
      for (int ni = 0; ni < 4; ++ni)
        acc[mi][ni] = __builtin_amdgcn_wmma_f32_16x16x32_f16(
            false, a[mi], false, b[ni], (short)0, acc[mi][ni], false, false);

    __syncthreads();   // protect buffer before next iteration overwrites it
  }

  float*  D32 = (float*) Dv + (size_t)z * (size_t)sD;
  half_t* D16 = (half_t*)Dv + (size_t)z * (size_t)sD;
#pragma unroll
  for (int ni = 0; ni < 4; ++ni) {
    int n = baseN + wn + ni * 16 + r;
    if (n >= Nn) continue;
    float bv = bias ? bias[n] : 0.0f;
#pragma unroll
    for (int mi = 0; mi < 2; ++mi) {
      int mbase = baseM + wm + mi * 16 + h2 * 8;  // D lane: col=r, rows mbase..+7
#pragma unroll
      for (int i = 0; i < 8; ++i) {
        int m = mbase + i;
        if (m >= M) continue;
        float v = acc[mi][ni][i] * alpha + bv;
        size_t off = (size_t)m * ldd + n;
        if (mode == EPI_F32) {
          D32[off] = v;
        } else if (mode == EPI_F16) {
          D16[off] = (half_t)v;
        } else if (mode == EPI_F16_GELU) {
          float g = 0.5f * v * (1.0f + erff(v * 0.70710678118654752f));
          D16[off] = (half_t)g;
        } else {                               // EPI_F32_BIAS_RES
          D32[off] = v + resid[off];
        }
      }
    }
  }
}

// ---------------------------------------------------------------------------
// Elementwise / reduction kernels
// ---------------------------------------------------------------------------
__global__ __launch_bounds__(256)
void layernorm_f16(const float* __restrict__ x, const float* __restrict__ g,
                   const float* __restrict__ bta, half_t* __restrict__ out)
{
  __shared__ float red[256];
  const int row = blockIdx.x, tid = threadIdx.x;
  const float* xr = x + (size_t)row * DIM_;
  float v0 = xr[tid], v1 = xr[tid + 256];
  red[tid] = v0 + v1; __syncthreads();
  for (int st = 128; st > 0; st >>= 1) {
    if (tid < st) red[tid] += red[tid + st];
    __syncthreads();
  }
  float mean = red[0] * (1.0f / DIM_); __syncthreads();
  float d0 = v0 - mean, d1 = v1 - mean;
  red[tid] = d0 * d0 + d1 * d1; __syncthreads();
  for (int st = 128; st > 0; st >>= 1) {
    if (tid < st) red[tid] += red[tid + st];
    __syncthreads();
  }
  float rstd = rsqrtf(red[0] * (1.0f / DIM_) + 1e-5f);
  half_t* orow = out + (size_t)row * DIM_;
  orow[tid]       = (half_t)(d0 * rstd * g[tid]       + bta[tid]);
  orow[tid + 256] = (half_t)(d1 * rstd * g[tid + 256] + bta[tid + 256]);
}

__global__ __launch_bounds__(256)
void softmax_f16(const float* __restrict__ dots, half_t* __restrict__ attn)
{
  __shared__ float red[256];
  const int row = blockIdx.x, tid = threadIdx.x;
  const float* dr = dots + (size_t)row * NPAD;
  float v = (tid < N_TOK) ? dr[tid] : -3.4e38f;
  red[tid] = v; __syncthreads();
  for (int st = 128; st > 0; st >>= 1) {
    if (tid < st) red[tid] = fmaxf(red[tid], red[tid + st]);
    __syncthreads();
  }
  float mx = red[0]; __syncthreads();
  float e = (tid < N_TOK) ? expf(v - mx) : 0.0f;
  red[tid] = e; __syncthreads();
  for (int st = 128; st > 0; st >>= 1) {
    if (tid < st) red[tid] += red[tid + st];
    __syncthreads();
  }
  float inv = 1.0f / red[0];
  if (tid < NPAD) attn[(size_t)row * NPAD + tid] = (half_t)(e * inv);
}

__device__ __forceinline__ float softplusf(float x) {
  return fmaxf(x, 0.0f) + log1pf(expf(-fabsf(x)));
}

__global__ __launch_bounds__(256)
void pos_embed(const float* __restrict__ Anoise, const float* __restrict__ bnoise,
               const float* __restrict__ Amean,  const float* __restrict__ bmean,
               const float* __restrict__ Astd,   const float* __restrict__ bstd,
               const float* __restrict__ X,      const float* __restrict__ Wfreq,
               half_t* __restrict__ pos)
{
  const int bp = blockIdx.x;
  const int b = bp / N_TOK, p = bp % N_TOK;
  const int o = threadIdx.x;                 // 0..255 frequencies
  float A00 = Amean[0] + softplusf(Astd[0]) * Anoise[b * 4 + 0];
  float A01 = Amean[1] + softplusf(Astd[1]) * Anoise[b * 4 + 1];
  float A10 = Amean[2] + softplusf(Astd[2]) * Anoise[b * 4 + 2];
  float A11 = Amean[3] + softplusf(Astd[3]) * Anoise[b * 4 + 3];
  float bv0 = bmean[0] + softplusf(bstd[0]) * bnoise[b * 2 + 0];
  float bv1 = bmean[1] + softplusf(bstd[1]) * bnoise[b * 2 + 1];
  float x0 = X[p * 2 + 0], x1 = X[p * 2 + 1];
  float t0 = x0 * A00 + x1 * A10 + bv0;      // X_t[b,p,o]=sum_d X[p,d]*A[d,o]
  float t1 = x0 * A01 + x1 * A11 + bv1;
  float pr = Wfreq[o * 2 + 0] * t0 + Wfreq[o * 2 + 1] * t1;
  const float s = 0.08838834764831845f;      // sqrt(2/256)
  half_t* pp = pos + (size_t)bp * DIM_;
  pp[o]       = (half_t)(s * cosf(pr));
  pp[o + 256] = (half_t)(s * sinf(pr));
}

__global__ __launch_bounds__(256)
void patchify_f16(const float* __restrict__ img, half_t* __restrict__ Ap)
{
  int idx = blockIdx.x * 256 + threadIdx.x;  // M_ALL*PD_ total, exact
  int k = idx % PD_, m = idx / PD_;
  int b = m / N_TOK, t = m % N_TOK;
  int gh = t / 14, gw = t % 14;
  int c = k % 3, q = k / 3;
  int p2 = q % 16, p1 = q / 16;
  Ap[idx] = (half_t)img[(((size_t)(b * 3 + c)) * 224 + gh * 16 + p1) * 224 +
                        gw * 16 + p2];
}

__global__ __launch_bounds__(256)
void unpatchify_f32(const float* __restrict__ y, float* __restrict__ out)
{
  int idx = blockIdx.x * 256 + threadIdx.x;  // BATCH*3*224*224 total, exact
  int b  = idx / (3 * 224 * 224);
  int rr = idx % (3 * 224 * 224);
  int c  = rr / (224 * 224);
  int r2 = rr % (224 * 224);
  int H = r2 / 224, W = r2 % 224;
  int gh = H / 16, p1 = H % 16, gw = W / 16, p2 = W % 16;
  int m = b * N_TOK + gh * 14 + gw;
  int k = (p1 * 16 + p2) * 3 + c;
  out[idx] = y[(size_t)m * PD_ + k];
}

__global__ __launch_bounds__(256)
void transpose_f32_f16(const float* __restrict__ src, half_t* __restrict__ dst,
                       int R, int Cc, int ld, int col0)
{
  int idx = blockIdx.x * 256 + threadIdx.x;
  if (idx >= R * Cc) return;
  int c = idx / R, rr = idx % R;
  dst[idx] = (half_t)src[(size_t)rr * ld + col0 + c];  // dst:(Cc,R)
}

__global__ __launch_bounds__(256)
void v_transpose(const half_t* __restrict__ v, half_t* __restrict__ vt)
{
  int idx = blockIdx.x * 256 + threadIdx.x;  // BATCH*DIM_*NPAD total, exact
  int b = idx / (DIM_ * NPAD);
  int q = idx % (DIM_ * NPAD);
  int n = q / NPAD, m = q % NPAD;
  half_t val = (half_t)0.0f;
  if (m < N_TOK) val = v[((size_t)(b * N_TOK + m)) * DIM_ + n];
  vt[idx] = val;
}

__global__ __launch_bounds__(256)
void f32_to_f16(const float* __restrict__ in, half_t* __restrict__ out, int n)
{
  int i = blockIdx.x * 256 + threadIdx.x;
  if (i < n) out[i] = (half_t)in[i];
}

// ---------------------------------------------------------------------------
// Host side
// ---------------------------------------------------------------------------
static inline void* wsp(void* base, size_t& off, size_t bytes) {
  void* p = (char*)base + off;
  off += (bytes + 255) & ~(size_t)255;
  return p;
}

static void launch_gemm(hipStream_t s,
                        const half_t* A, long long sA, int lda,
                        const half_t* Bt, long long sB, int ldb,
                        void* D, long long sD, int ldd,
                        const float* bias, const float* resid,
                        int M, int Nn, int K, float alpha, int mode, int batches)
{
  dim3 grid((M + 63) / 64, (Nn + 127) / 128, batches);
  gemm_wmma<<<grid, 128, 0, s>>>(A, sA, lda, Bt, sB, ldb, D, sD, ldd,
                                 bias, resid, M, Nn, K, alpha, mode);
}

extern "C" void kernel_launch(void* const* d_in, const int* in_sizes, int n_in,
                              void* d_out, int out_size, void* d_ws, size_t ws_size,
                              hipStream_t stream) {
  (void)in_sizes; (void)n_in; (void)out_size; (void)ws_size;
  const float* img     = (const float*)d_in[0];
  const float* A_noise = (const float*)d_in[2];
  const float* b_noise = (const float*)d_in[3];
  const float* Wfreq   = (const float*)d_in[4];
  const float* X       = (const float*)d_in[5];
  const float* basepos = (const float*)d_in[6];
  const float* A_mean  = (const float*)d_in[7];
  const float* b_mean  = (const float*)d_in[8];
  const float* A_std   = (const float*)d_in[9];
  const float* b_std   = (const float*)d_in[10];
  const float* Wpe     = (const float*)d_in[11];
  const float* bpe     = (const float*)d_in[12];
  const float* Wep     = (const float*)d_in[13];
  const float* bep     = (const float*)d_in[14];
  const float* ln1_g   = (const float*)d_in[15];
  const float* ln1_b   = (const float*)d_in[16];
  const float* Wqkv    = (const float*)d_in[17];
  const float* Wout    = (const float*)d_in[18];
  const float* bout    = (const float*)d_in[19];
  const float* ln2_g   = (const float*)d_in[20];
  const float* ln2_b   = (const float*)d_in[21];
  const float* Wff1    = (const float*)d_in[22];
  const float* bff1    = (const float*)d_in[23];
  const float* Wff2    = (const float*)d_in[24];
  const float* bff2    = (const float*)d_in[25];

  size_t off = 0;
  half_t* wpeT  = (half_t*)wsp(d_ws, off, (size_t)DIM_ * PD_ * 2);
  half_t* wvT   = (half_t*)wsp(d_ws, off, (size_t)4 * DIM_ * DIM_ * 2);
  half_t* woutT = (half_t*)wsp(d_ws, off, (size_t)4 * DIM_ * DIM_ * 2);
  half_t* wff1T = (half_t*)wsp(d_ws, off, (size_t)4 * MLP_ * DIM_ * 2);
  half_t* wff2T = (half_t*)wsp(d_ws, off, (size_t)4 * DIM_ * MLP_ * 2);
  half_t* wepT  = (half_t*)wsp(d_ws, off, (size_t)PD_ * DIM_ * 2);
  half_t* bpos  = (half_t*)wsp(d_ws, off, (size_t)N_TOK * DIM_ * 2);
  float*  xbuf  = (float*) wsp(d_ws, off, (size_t)M_ALL * DIM_ * 4);
  half_t* xn    = (half_t*)wsp(d_ws, off, (size_t)M_ALL * DIM_ * 2);
  half_t* posv  = (half_t*)wsp(d_ws, off, (size_t)M_ALL * DIM_ * 2);   // pos | v
  char*   dvt   = (char*)  wsp(d_ws, off, (size_t)BATCH * DIM_ * NPAD * 2); // dots | v^T
  half_t* attn  = (half_t*)wsp(d_ws, off, (size_t)M_ALL * NPAD * 2);
  half_t* out16 = (half_t*)wsp(d_ws, off, (size_t)M_ALL * DIM_ * 2);
  char*   big   = (char*)  wsp(d_ws, off, (size_t)M_ALL * MLP_ * 2);   // Apatch | h | y

  // ---- per-call weight prep: f32 -> f16, transposed to (N,K) ----
  auto T = [&](const float* src, half_t* dst, int R, int Cc, int ld, int c0) {
    int tot = R * Cc;
    transpose_f32_f16<<<(tot + 255) / 256, 256, 0, stream>>>(src, dst, R, Cc, ld, c0);
  };
  T(Wpe, wpeT, PD_, DIM_, DIM_, 0);
  for (int l = 0; l < 4; ++l) {
    T(Wqkv + (size_t)l * DIM_ * 3 * DIM_, wvT  + (size_t)l * DIM_ * DIM_,
      DIM_, DIM_, 3 * DIM_, 2 * DIM_);                 // V slice only
    T(Wout + (size_t)l * DIM_ * DIM_,  woutT + (size_t)l * DIM_ * DIM_,
      DIM_, DIM_, DIM_, 0);
    T(Wff1 + (size_t)l * DIM_ * MLP_,  wff1T + (size_t)l * MLP_ * DIM_,
      DIM_, MLP_, MLP_, 0);
    T(Wff2 + (size_t)l * MLP_ * DIM_,  wff2T + (size_t)l * DIM_ * MLP_,
      MLP_, DIM_, DIM_, 0);
  }
  T(Wep, wepT, DIM_, PD_, PD_, 0);
  f32_to_f16<<<(N_TOK * DIM_ + 255) / 256, 256, 0, stream>>>(basepos, bpos, N_TOK * DIM_);

  // ---- patchify + patch embedding ----
  patchify_f16<<<(M_ALL * PD_) / 256, 256, 0, stream>>>(img, (half_t*)big);
  launch_gemm(stream, (half_t*)big, 0, PD_, wpeT, 0, PD_, xbuf, 0, DIM_,
              bpe, nullptr, M_ALL, DIM_, PD_, 1.0f, EPI_F32, 1);

  // ---- transformed positional embedding + attention scores ----
  pos_embed<<<M_ALL, 256, 0, stream>>>(A_noise, b_noise, A_mean, b_mean,
                                       A_std, b_std, X, Wfreq, posv);
  const float SCALE = 0.044194173824159216f;  // 512^-0.5
  launch_gemm(stream, posv, 0, DIM_, bpos, 0, DIM_, (float*)dvt, 0, NPAD,
              nullptr, nullptr, M_ALL, N_TOK, DIM_, SCALE, EPI_F32, 1);
  softmax_f16<<<M_ALL, 256, 0, stream>>>((float*)dvt, attn);

  // ---- transformer layers ----
  for (int l = 0; l < 4; ++l) {
    layernorm_f16<<<M_ALL, 256, 0, stream>>>(xbuf, ln1_g + l * DIM_,
                                             ln1_b + l * DIM_, xn);
    // v = xn @ Wv            (f16 out)
    launch_gemm(stream, xn, 0, DIM_, wvT + (size_t)l * DIM_ * DIM_, 0, DIM_,
                posv, 0, DIM_, nullptr, nullptr, M_ALL, DIM_, DIM_, 1.0f, EPI_F16, 1);
    // v^T (B, DIM, NPAD) for contiguous B-fragments in attn@v
    v_transpose<<<(BATCH * DIM_ * NPAD) / 256, 256, 0, stream>>>(posv, (half_t*)dvt);
    // out = attn @ v   (per batch, K padded to 224 with zero attn weights)
    launch_gemm(stream, attn, (long long)N_TOK * NPAD, NPAD,
                (half_t*)dvt, (long long)DIM_ * NPAD, NPAD,
                out16, (long long)N_TOK * DIM_, DIM_,
                nullptr, nullptr, N_TOK, DIM_, NPAD, 1.0f, EPI_F16, BATCH);
    // x = out @ Wout + bout  (NO residual around attention)
    launch_gemm(stream, out16, 0, DIM_, woutT + (size_t)l * DIM_ * DIM_, 0, DIM_,
                xbuf, 0, DIM_, bout + l * DIM_, nullptr,
                M_ALL, DIM_, DIM_, 1.0f, EPI_F32, 1);
    // residual MLP
    layernorm_f16<<<M_ALL, 256, 0, stream>>>(xbuf, ln2_g + l * DIM_,
                                             ln2_b + l * DIM_, xn);
    launch_gemm(stream, xn, 0, DIM_, wff1T + (size_t)l * MLP_ * DIM_, 0, DIM_,
                (half_t*)big, 0, MLP_, bff1 + l * MLP_, nullptr,
                M_ALL, MLP_, DIM_, 1.0f, EPI_F16_GELU, 1);
    launch_gemm(stream, (half_t*)big, 0, MLP_, wff2T + (size_t)l * DIM_ * MLP_, 0, MLP_,
                xbuf, 0, DIM_, bff2 + l * DIM_, xbuf,
                M_ALL, DIM_, MLP_, 1.0f, EPI_F32_BIAS_RES, 1);
  }

  // ---- embedding_to_patch + unpatchify ----
  f32_to_f16<<<(M_ALL * DIM_ + 255) / 256, 256, 0, stream>>>(xbuf, xn, M_ALL * DIM_);
  launch_gemm(stream, xn, 0, DIM_, wepT, 0, DIM_, (float*)big, 0, PD_,
              bep, nullptr, M_ALL, PD_, DIM_, 1.0f, EPI_F32, 1);
  unpatchify_f32<<<(BATCH * 3 * 224 * 224) / 256, 256, 0, stream>>>(
      (float*)big, (float*)d_out);
}
